// WindowAttentionPseudo3d_56624848830779
// MI455X (gfx1250) — matrix operand order repaired and checked by reading, not compiled, not run on hardware
//
#include <hip/hip_runtime.h>

typedef _Float16 f16;
typedef __attribute__((ext_vector_type(16))) _Float16 v16h;
typedef __attribute__((ext_vector_type(8)))  _Float16 v8h;
typedef __attribute__((ext_vector_type(8)))  float    v8f;

#define DIMC 128
#define HEADS 4
#define HD 32
#define LTOK 256
#define BWIN 512
#define LOG100 4.605170185988091f

// Byte offset of the Xh/S union region inside the dynamic-LDS block (Qh,Kh,VT precede it).
#define XH_LDS_BYTE_OFF (3 * 65536)

// ---------------- WMMA fragment helpers (CDNA5 16x16x32 f16 layouts) ----------------
// A (16x32, f16): lane&15 = row M; lanes<16 hold K=0..15, lanes>=16 hold K=16..31.
__device__ __forceinline__ v16h afrag_lds(const f16* base, int ld, int row0, int lane) {
  const f16* pr = base + (row0 + (lane & 15)) * ld + ((lane >> 4) << 4);
  v8h lo = *(const v8h*)pr;
  v8h hi = *(const v8h*)(pr + 8);
  v16h a;
#pragma unroll
  for (int i = 0; i < 8; ++i) { a[i] = lo[i]; a[i + 8] = hi[i]; }
  return a;
}

// B (32x16, f16): lane&15 = column N; lanes<16 hold K=0..15, lanes>=16 hold K=16..31.
// "base" is laid out so that B-column c is 'base[c*ld + k]' (rows of K / rows of W / rows of V^T).
__device__ __forceinline__ v16h bfrag_lds(const f16* base, int ld, int col0, int lane) {
  const f16* pc = base + (col0 + (lane & 15)) * ld + ((lane >> 4) << 4);
  v8h lo = *(const v8h*)pc;
  v8h hi = *(const v8h*)(pc + 8);
  v16h b;
#pragma unroll
  for (int i = 0; i < 8; ++i) { b[i] = lo[i]; b[i + 8] = hi[i]; }
  return b;
}

// B fragment from pre-converted f16 global weights W16[n][k] (row-major, ld=K): two b128 loads, no cvt.
__device__ __forceinline__ v16h bfrag_gf16(const f16* __restrict__ w, int ld, int col0, int k0, int lane) {
  const f16* p = w + (size_t)(col0 + (lane & 15)) * ld + k0 + ((lane >> 4) << 4);
  v8h lo = *(const v8h*)p;
  v8h hi = *(const v8h*)(p + 8);
  v16h b;
#pragma unroll
  for (int i = 0; i < 8; ++i) { b[i] = lo[i]; b[i + 8] = hi[i]; }
  return b;
}

__device__ __forceinline__ v8f wmma_f16(v16h a, v16h b, v8f c) {
  return __builtin_amdgcn_wmma_f32_16x16x32_f16(false, a, false, b, (short)0, c, false, false);
}

// ---------------- Prep: f32 -> f16 conversion (weights + activations), done once ----------------
extern "C" __global__ void f32_to_f16_kernel(const float* __restrict__ src,
                                             f16* __restrict__ dst, int n4) {
  int i = blockIdx.x * blockDim.x + threadIdx.x;
  if (i >= n4) return;
  float4 v = ((const float4*)src)[i];
  dst[i * 4 + 0] = (f16)v.x;
  dst[i * 4 + 1] = (f16)v.y;
  dst[i * 4 + 2] = (f16)v.z;
  dst[i * 4 + 3] = (f16)v.w;
}

// ---------------- CPB MLP: bias_table[m][h] = (relu(table@w1^T + b1) @ w2^T)[m][h] ----------------
extern "C" __global__ void cpb_mlp_kernel(const float* __restrict__ rel_table,
                                          const float* __restrict__ w1,
                                          const float* __restrict__ b1,
                                          const float* __restrict__ w2,
                                          float* __restrict__ bias_table, int M) {
  int m = blockIdx.x * blockDim.x + threadIdx.x;
  if (m >= M) return;
  float t0 = rel_table[m * 3 + 0];
  float t1 = rel_table[m * 3 + 1];
  float t2 = rel_table[m * 3 + 2];
  float acc0 = 0.f, acc1 = 0.f, acc2 = 0.f, acc3 = 0.f;
  for (int j = 0; j < 512; ++j) {
    float hsum = t0 * w1[j * 3 + 0] + t1 * w1[j * 3 + 1] + t2 * w1[j * 3 + 2] + b1[j];
    hsum = fmaxf(hsum, 0.f);
    acc0 += hsum * w2[0 * 512 + j];
    acc1 += hsum * w2[1 * 512 + j];
    acc2 += hsum * w2[2 * 512 + j];
    acc3 += hsum * w2[3 * 512 + j];
  }
  bias_table[m * HEADS + 0] = acc0;
  bias_table[m * HEADS + 1] = acc1;
  bias_table[m * HEADS + 2] = acc2;
  bias_table[m * HEADS + 3] = acc3;
}

// biasAdd[h][i][j] = 16*sigmoid(bias_table[rel_index[i][j]][h])
extern "C" __global__ void cpb_gather_kernel(const int* __restrict__ rel_index,
                                             const float* __restrict__ bias_table,
                                             float* __restrict__ biasAdd) {
  int t = blockIdx.x * blockDim.x + threadIdx.x;  // 0 .. L*L-1
  int idx = rel_index[t];
#pragma unroll
  for (int h = 0; h < HEADS; ++h) {
    float v = bias_table[idx * HEADS + h];
    biasAdd[(size_t)h * LTOK * LTOK + t] = 16.f / (1.f + __expf(-v));
  }
}

// ---------------- Fused window attention: one workgroup per window ----------------
extern "C" __global__ __launch_bounds__(256, 1)
void win_attn_fused_kernel(const f16* __restrict__ x16,
                           const f16* __restrict__ qkv_w16,
                           const float* __restrict__ q_bias,
                           const float* __restrict__ v_bias,
                           const float* __restrict__ logit_scale,
                           const f16* __restrict__ proj_w16,
                           const float* __restrict__ proj_b,
                           const float* __restrict__ biasAdd,
                           float* __restrict__ out) {
  extern __shared__ __align__(16) char smem[];
  f16* Qh = (f16*)smem;        // 256 x 128 f16  (64 KB)
  f16* Kh = Qh + 32768;        // 256 x 128 f16  (64 KB)
  f16* VT = Kh + 32768;        // 128 x 256 f16  (V transposed, 64 KB)
  f16* SX = VT + 32768;        // union: Xh 256x128 | S 128x256 (64 KB)
  f16* Oh = SX + 32768;        // 256 x 128 f16  (64 KB)   -> 320 KB total

  const int tid  = threadIdx.x;
  const int lane = tid & 31;
  const int w    = tid >> 5;   // wave id 0..7
  const int bwin = blockIdx.x; // window id

  // ---- Phase 0: async-copy the f16 x tile (64 KB) straight into LDS (Xh region) ----
  // GLOBAL_LOAD_ASYNC_TO_LDS_B128: no VGPR data return, tracked by ASYNCcnt.
  {
    const f16* xw16 = x16 + (size_t)bwin * LTOK * DIMC;
#pragma unroll
    for (int it = 0; it < 16; ++it) {
      int chunk = it * 256 + tid;                         // 4096 x 16-byte chunks
      unsigned ldsoff = XH_LDS_BYTE_OFF + chunk * 16;     // LDS byte offset (VDST)
      const f16* g = xw16 + chunk * 8;                    // global address (VADDR)
      asm volatile("global_load_async_to_lds_b128 %0, %1, off"
                   :: "v"(ldsoff), "v"(g) : "memory");
    }
    asm volatile("s_wait_asynccnt 0x0" ::: "memory");
  }
  __syncthreads();
  f16* Xh = SX;

  // ---- Phase 1: QKV GEMM (256x128)@(128x384), write Qh/Kh/VT as f16 (+bias) ----
  for (int nt = 0; nt < 24; ++nt) {
    v16h b0 = bfrag_gf16(qkv_w16, DIMC, nt * 16, 0, lane);
    v16h b1 = bfrag_gf16(qkv_w16, DIMC, nt * 16, 32, lane);
    v16h b2 = bfrag_gf16(qkv_w16, DIMC, nt * 16, 64, lane);
    v16h b3 = bfrag_gf16(qkv_w16, DIMC, nt * 16, 96, lane);
    const int ncol = nt * 16 + (lane & 15);  // 0..383
#pragma unroll
    for (int mi = 0; mi < 2; ++mi) {
      const int m0 = (w * 2 + mi) * 16;
      v8f acc = {};
      acc = wmma_f16(afrag_lds(Xh + 0,  DIMC, m0, lane), b0, acc);
      acc = wmma_f16(afrag_lds(Xh + 32, DIMC, m0, lane), b1, acc);
      acc = wmma_f16(afrag_lds(Xh + 64, DIMC, m0, lane), b2, acc);
      acc = wmma_f16(afrag_lds(Xh + 96, DIMC, m0, lane), b3, acc);
      const int rbase = m0 + ((lane >> 4) << 3);
      if (ncol < 128) {               // Q (+q_bias)
        float qb = q_bias[ncol];
#pragma unroll
        for (int i = 0; i < 8; ++i) Qh[(rbase + i) * DIMC + ncol] = (f16)(acc[i] + qb);
      } else if (ncol < 256) {        // K (bias = 0)
        int c = ncol - 128;
#pragma unroll
        for (int i = 0; i < 8; ++i) Kh[(rbase + i) * DIMC + c] = (f16)acc[i];
      } else {                        // V (+v_bias), stored transposed
        int c = ncol - 256;
        float vb = v_bias[c];
#pragma unroll
        for (int i = 0; i < 8; ++i) VT[c * LTOK + (rbase + i)] = (f16)(acc[i] + vb);
      }
    }
  }
  __syncthreads();

  // ---- Phase 2: cosine normalization of q,k per (row, head) ----
  {
    const int r = tid;  // 256 threads, 256 rows
#pragma unroll
    for (int h = 0; h < HEADS; ++h) {
      f16* q = Qh + r * DIMC + h * HD;
      f16* k = Kh + r * DIMC + h * HD;
      float ssq = 0.f, ssk = 0.f;
#pragma unroll 8
      for (int j = 0; j < HD; ++j) { float a = (float)q[j]; ssq += a * a; float b = (float)k[j]; ssk += b * b; }
      float iq = 1.f / fmaxf(sqrtf(ssq), 1e-12f);
      float ik = 1.f / fmaxf(sqrtf(ssk), 1e-12f);
#pragma unroll 8
      for (int j = 0; j < HD; ++j) { q[j] = (f16)((float)q[j] * iq); k[j] = (f16)((float)k[j] * ik); }
    }
  }
  __syncthreads();

  // ---- Phase 3: attention, per head, in two 128-row blocks (S lives in SX) ----
  f16* S = SX;  // 128 x 256 f16
  for (int h = 0; h < HEADS; ++h) {
    const float scale = __expf(fminf(logit_scale[h], LOG100));
    const float* bA = biasAdd + (size_t)h * LTOK * LTOK;
    for (int rb = 0; rb < 2; ++rb) {
      // S = Qn @ Kn^T : one WMMA per 16x16 tile (K = 32 = head dim)
      const int rloc0 = w * 16;                 // local row tile of this wave
      const int rg0   = rb * 128 + rloc0;       // global query row base
      v16h aq = afrag_lds(Qh + h * HD, DIMC, rg0, lane);
      for (int nt = 0; nt < 16; ++nt) {
        v16h bk = bfrag_lds(Kh + h * HD, DIMC, nt * 16, lane);
        v8f acc = {};
        acc = wmma_f16(aq, bk, acc);
        const int ncol  = nt * 16 + (lane & 15);
        const int rbase = ((lane >> 4) << 3);
#pragma unroll
        for (int i = 0; i < 8; ++i) {
          float sv = acc[i] * scale + bA[(size_t)(rg0 + rbase + i) * LTOK + ncol];
          S[(rloc0 + rbase + i) * LTOK + ncol] = (f16)sv;
        }
      }
      __syncthreads();

      // softmax over each of the 128 rows (one thread per row)
      if (tid < 128) {
        f16* row = S + tid * LTOK;
        float m = -1e30f;
        for (int j = 0; j < LTOK; ++j) m = fmaxf(m, (float)row[j]);
        float s = 0.f;
        for (int j = 0; j < LTOK; ++j) { float e = __expf((float)row[j] - m); s += e; row[j] = (f16)e; }
        float inv = 1.f / s;
        for (int j = 0; j < LTOK; ++j) row[j] = (f16)((float)row[j] * inv);
      }
      __syncthreads();

      // O = P @ V : M=128 (one 16-row tile per wave), N=32, K=256
      {
        v8f o0 = {}, o1 = {};
#pragma unroll
        for (int ks = 0; ks < 8; ++ks) {
          v16h ap  = afrag_lds(S + ks * 32, LTOK, rloc0, lane);
          v16h bv0 = bfrag_lds(VT + (size_t)(h * HD) * LTOK + ks * 32, LTOK, 0,  lane);
          v16h bv1 = bfrag_lds(VT + (size_t)(h * HD) * LTOK + ks * 32, LTOK, 16, lane);
          o0 = wmma_f16(ap, bv0, o0);
          o1 = wmma_f16(ap, bv1, o1);
        }
        const int rbase = rb * 128 + rloc0 + ((lane >> 4) << 3);
        const int c0 = h * HD + (lane & 15);
#pragma unroll
        for (int i = 0; i < 8; ++i) {
          Oh[(rbase + i) * DIMC + c0]      = (f16)o0[i];
          Oh[(rbase + i) * DIMC + c0 + 16] = (f16)o1[i];
        }
      }
      __syncthreads();
    }
  }

  // ---- Phase 4: output projection (256x128)@(128x128) + bias -> f32 out ----
  float* outw = out + (size_t)bwin * LTOK * DIMC;
  for (int nt = 0; nt < 8; ++nt) {
    v16h p0 = bfrag_gf16(proj_w16, DIMC, nt * 16, 0, lane);
    v16h p1 = bfrag_gf16(proj_w16, DIMC, nt * 16, 32, lane);
    v16h p2 = bfrag_gf16(proj_w16, DIMC, nt * 16, 64, lane);
    v16h p3 = bfrag_gf16(proj_w16, DIMC, nt * 16, 96, lane);
    const int ncol = nt * 16 + (lane & 15);
    const float pb = proj_b[ncol];
#pragma unroll
    for (int mi = 0; mi < 2; ++mi) {
      const int m0 = (w * 2 + mi) * 16;
      v8f acc = {};
      acc = wmma_f16(afrag_lds(Oh + 0,  DIMC, m0, lane), p0, acc);
      acc = wmma_f16(afrag_lds(Oh + 32, DIMC, m0, lane), p1, acc);
      acc = wmma_f16(afrag_lds(Oh + 64, DIMC, m0, lane), p2, acc);
      acc = wmma_f16(afrag_lds(Oh + 96, DIMC, m0, lane), p3, acc);
      const int rbase = m0 + ((lane >> 4) << 3);
#pragma unroll
      for (int i = 0; i < 8; ++i)
        outw[(size_t)(rbase + i) * DIMC + ncol] = acc[i] + pb;
    }
  }
}

extern "C" void kernel_launch(void* const* d_in, const int* in_sizes, int n_in,
                              void* d_out, int out_size, void* d_ws, size_t ws_size,
                              hipStream_t stream) {
  const float* x           = (const float*)d_in[0];
  const float* qkv_w       = (const float*)d_in[1];
  const float* q_bias      = (const float*)d_in[2];
  const float* v_bias      = (const float*)d_in[3];
  const float* logit_scale = (const float*)d_in[4];
  const float* cpb_w1      = (const float*)d_in[5];
  const float* cpb_b1      = (const float*)d_in[6];
  const float* cpb_w2      = (const float*)d_in[7];
  const float* proj_w      = (const float*)d_in[8];
  const float* proj_b      = (const float*)d_in[9];
  const float* rel_table   = (const float*)d_in[10];
  const int*   rel_index   = (const int*)d_in[11];
  float* out = (float*)d_out;

  const int M = in_sizes[10] / 3;  // 1575 relative-coordinate rows

  // Workspace carve-up (byte offsets, 256-aligned):
  char* ws = (char*)d_ws;
  float* bias_table = (float*)(ws + 0);              //  1575 x 4 f32   (<32 KB)
  float* biasAdd    = (float*)(ws + 32768);          //  4 x 256 x 256 f32 (1 MB)
  f16*   qkv_w16    = (f16*)  (ws + 1081344);        //  384 x 128 f16  (96 KB)
  f16*   proj_w16   = (f16*)  (ws + 1179648);        //  128 x 128 f16  (32 KB)
  f16*   x16        = (f16*)  (ws + 1212416);        //  512*256*128 f16 (32 MB)

  // One-time f32 -> f16 conversions (weights + activations).
  f32_to_f16_kernel<<<(3 * DIMC * DIMC / 4 + 255) / 256, 256, 0, stream>>>(qkv_w, qkv_w16,
                                                                           3 * DIMC * DIMC / 4);
  f32_to_f16_kernel<<<(DIMC * DIMC / 4 + 255) / 256, 256, 0, stream>>>(proj_w, proj_w16,
                                                                       DIMC * DIMC / 4);
  const int xn4 = BWIN * LTOK * DIMC / 4;
  f32_to_f16_kernel<<<(xn4 + 255) / 256, 256, 0, stream>>>(x, x16, xn4);

  // Relative-position bias (tiny).
  cpb_mlp_kernel<<<(M + 127) / 128, 128, 0, stream>>>(rel_table, cpb_w1, cpb_b1, cpb_w2,
                                                      bias_table, M);
  cpb_gather_kernel<<<(LTOK * LTOK) / 256, 256, 0, stream>>>(rel_index, bias_table, biasAdd);

  const size_t lds_bytes = 5 * 65536;  // 320 KB: Qh + Kh + VT + (Xh|S) + Oh
  win_attn_fused_kernel<<<BWIN, 256, lds_bytes, stream>>>(x16, qkv_w16, q_bias, v_bias,
                                                          logit_scale, proj_w16, proj_b,
                                                          biasAdd, out);
}
